// CrossAttention_Single_48971217109024
// MI455X (gfx1250) — matrix operand order repaired and checked by reading, not compiled, not run on hardware
//
#include <hip/hip_runtime.h>

// ---------------------------------------------------------------------------
// CrossAttention (channel attention / XCA style) for MI455X (gfx1250, wave32)
// All GEMMs run on v_wmma_f32_16x16x32_bf16 (f32 accumulate, bf16 operands).
// Each wave computes a 32(M) x 64(N) tile: 8 WMMAs per 32-wide K step,
// 1.5 global_load_b128 per WMMA (A fragments reused 4x, B fragments 2x).
// ---------------------------------------------------------------------------

typedef __attribute__((ext_vector_type(16))) __bf16 v16bf;
typedef __attribute__((ext_vector_type(8)))  __bf16 v8bf;
typedef __attribute__((ext_vector_type(8)))  float  v8f;

#define BATCH  32
#define NTOK   256
#define CDIM   4096
#define NHEAD  16
#define HD     256
#define MROWS  (BATCH * NTOK)          // 8192
#define NELEM_X ((size_t)MROWS * CDIM) // 33554432
#define NELEM_W ((size_t)CDIM * CDIM)  // 16777216

// ---------------- K0: f32 -> bf16 elementwise convert ----------------------
__global__ void cvt_f32_bf16(const float* __restrict__ src,
                             __bf16* __restrict__ dst, int n) {
  int i = blockIdx.x * blockDim.x + threadIdx.x;
  if (i < n) dst[i] = (__bf16)src[i];
}

// ---------------- WMMA tile loaders ----------------------------------------
// A 16x32 bf16 tile from row-major [M][K] (lda in elements):
//   lanes 0-15: M=lane, K chunks [0..7] and [16..23]
//   lanes 16-31: M=lane-16, K chunks [8..15] and [24..31]
__device__ inline v16bf load_A(const __bf16* base, int lda, int lane) {
  const __bf16* p = base + (size_t)(lane & 15) * lda + ((lane >> 4) << 3);
  v8bf lo = *(const v8bf*)p;
  v8bf hi = *(const v8bf*)(p + 16);
  v16bf r;
#pragma unroll
  for (int t = 0; t < 8; ++t) { r[t] = lo[t]; r[t + 8] = hi[t]; }
  return r;
}

// B 32x16 bf16 tile from [N][K] (K-contiguous, ldb in elements):
//   lanes 0-15: N=lane, K chunks [0..7] and [16..23]; lanes 16-31: K chunks [8..15] and [24..31]
__device__ inline v16bf load_B(const __bf16* baseNT, int ldb, int lane) {
  const __bf16* p = baseNT + (size_t)(lane & 15) * ldb + ((lane >> 4) << 3);
  v8bf lo = *(const v8bf*)p;
  v8bf hi = *(const v8bf*)(p + 16);
  v16bf r;
#pragma unroll
  for (int t = 0; t < 8; ++t) { r[t] = lo[t]; r[t + 8] = hi[t]; }
  return r;
}

// One wave computes a 32(M) x 64(N) tile; acc[i*4+j] is (m-tile i, n-tile j).
__device__ inline void wave_gemm_32x64(const __bf16* __restrict__ A, int lda,
                                       const __bf16* __restrict__ Bt, int ldb,
                                       int K, int lane, v8f acc[8]) {
  for (int k0 = 0; k0 < K; k0 += 32) {
    __builtin_prefetch(A + k0 + 128, 0, 1);                 // global_prefetch_b8
    __builtin_prefetch(A + (size_t)16 * lda + k0 + 128, 0, 1);
    v16bf a0 = load_A(A + k0, lda, lane);
    v16bf a1 = load_A(A + (size_t)16 * lda + k0, lda, lane);
#pragma unroll
    for (int j = 0; j < 4; ++j) {
      v16bf b = load_B(Bt + (size_t)(j * 16) * ldb + k0, ldb, lane);
      acc[j]     = __builtin_amdgcn_wmma_f32_16x16x32_bf16(
          false, a0, false, b, (short)0, acc[j], false, false);
      acc[4 + j] = __builtin_amdgcn_wmma_f32_16x16x32_bf16(
          false, a1, false, b, (short)0, acc[4 + j], false, false);
    }
  }
}

// ---------------- K1: QKV projection, store transposed [B,H,HD,N] ----------
// out = (X[8192x4096] @ W^T) * scale, scattered to outT[b][h][chan][token]
__global__ void gemm_qkv_T(const __bf16* __restrict__ X,
                           const __bf16* __restrict__ W,   // [out_ch][in_ch]
                           __bf16* __restrict__ outT, float scale) {
  int lane = threadIdx.x;
  int waveId = blockIdx.x * 8 + threadIdx.y;   // 16384 wave-tiles
  int m0 = (waveId & 255) << 5;                // 256 M-tiles (32 rows)
  int n0 = (waveId >> 8) << 6;                 // 64 N-strips
  v8f acc[8] = {};
  wave_gemm_32x64(X + (size_t)m0 * CDIM, CDIM,
                  W + (size_t)n0 * CDIM, CDIM, CDIM, lane, acc);
  int half = lane >> 4, nl = lane & 15;
#pragma unroll
  for (int i = 0; i < 2; ++i)
#pragma unroll
    for (int j = 0; j < 4; ++j) {
      int o = n0 + j * 16 + nl;           // output channel
      int h = o >> 8, ic = o & 255;       // head, channel-in-head
#pragma unroll
      for (int r = 0; r < 8; ++r) {
        int mt = m0 + i * 16 + r + (half << 3);   // token row
        int b = mt >> 8, t = mt & 255;
        outT[((((size_t)b * NHEAD + h) * HD + ic) << 8) + t] =
            (__bf16)(acc[i * 4 + j][r] * scale);
      }
    }
}

// ---------------- K2: per-head logits = q^T k  (f32 -> d_out attn region) --
__global__ void gemm_head_logits(const __bf16* __restrict__ qT,
                                 const __bf16* __restrict__ kT,
                                 float* __restrict__ logits) {
  int lane = threadIdx.x;
  int head = blockIdx.x >> 2;                       // 512 heads, 4 blocks/head
  int tile = ((blockIdx.x & 3) << 3) + threadIdx.y; // 32 wave-tiles/head
  int m0 = (tile & 7) << 5;
  int n0 = (tile >> 3) << 6;
  const __bf16* A  = qT + (size_t)head * HD * NTOK + (size_t)m0 * NTOK;
  const __bf16* Bt = kT + (size_t)head * HD * NTOK + (size_t)n0 * NTOK;
  v8f acc[8] = {};
  wave_gemm_32x64(A, NTOK, Bt, NTOK, NTOK, lane, acc);
  float* C = logits + (size_t)head * HD * HD;
  int half = lane >> 4, nl = lane & 15;
#pragma unroll
  for (int i = 0; i < 2; ++i)
#pragma unroll
    for (int j = 0; j < 4; ++j)
#pragma unroll
      for (int r = 0; r < 8; ++r)
        C[(size_t)(m0 + i * 16 + r + (half << 3)) * HD + n0 + j * 16 + nl] =
            acc[i * 4 + j][r];
}

// ---------------- K3: in-place softmax, one wave per 256-wide row ----------
__global__ void softmax_rows(float* __restrict__ attn,
                             __bf16* __restrict__ attn_bf) {
  int lane = threadIdx.x;
  size_t row = (size_t)blockIdx.x * 8 + threadIdx.y;  // 131072 rows
  float* p = attn + row * 256;
  float v[8];
  float mx = -3.402823466e38f;
#pragma unroll
  for (int t = 0; t < 8; ++t) { v[t] = p[lane + t * 32]; mx = fmaxf(mx, v[t]); }
#pragma unroll
  for (int s = 16; s > 0; s >>= 1) mx = fmaxf(mx, __shfl_xor(mx, s, 32));
  float sum = 0.f;
#pragma unroll
  for (int t = 0; t < 8; ++t) { v[t] = __expf(v[t] - mx); sum += v[t]; }
#pragma unroll
  for (int s = 16; s > 0; s >>= 1) sum += __shfl_xor(sum, s, 32);
  float inv = 1.f / sum;
#pragma unroll
  for (int t = 0; t < 8; ++t) {
    float o = v[t] * inv;
    p[lane + t * 32] = o;
    attn_bf[row * 256 + lane + t * 32] = (__bf16)o;
  }
}

// ---------------- K4: per-head out = attn @ v  (bf16 row-major) ------------
__global__ void gemm_head_av(const __bf16* __restrict__ attn_bf,
                             const __bf16* __restrict__ vT,
                             __bf16* __restrict__ outA) {
  int lane = threadIdx.x;
  int head = blockIdx.x >> 2;
  int tile = ((blockIdx.x & 3) << 3) + threadIdx.y;
  int m0 = (tile & 7) << 5;
  int n0 = (tile >> 3) << 6;
  const __bf16* A  = attn_bf + (size_t)head * HD * NTOK + (size_t)m0 * NTOK;
  const __bf16* Bt = vT      + (size_t)head * HD * NTOK + (size_t)n0 * NTOK;
  v8f acc[8] = {};
  wave_gemm_32x64(A, NTOK, Bt, NTOK, NTOK, lane, acc);
  __bf16* C = outA + (size_t)head * HD * HD;
  int half = lane >> 4, nl = lane & 15;
#pragma unroll
  for (int i = 0; i < 2; ++i)
#pragma unroll
    for (int j = 0; j < 4; ++j)
#pragma unroll
      for (int r = 0; r < 8; ++r)
        C[(size_t)(m0 + i * 16 + r + (half << 3)) * HD + n0 + j * 16 + nl] =
            (__bf16)acc[i * 4 + j][r];
}

// ---------------- K5: final projection x = out @ Wp^T + bp  (f32) ----------
// A rows are final tokens: A(m, k=h*256+c) lives at outA[((b*16+h)*256+t)*256+c]
__global__ void gemm_proj(const __bf16* __restrict__ AO,
                          const __bf16* __restrict__ Wp,  // [o][c]
                          const float* __restrict__ bp,
                          float* __restrict__ X) {
  int lane = threadIdx.x;
  int waveId = blockIdx.x * 8 + threadIdx.y;
  int m0 = (waveId & 255) << 5;
  int n0 = (waveId >> 8) << 6;
  int b = m0 >> 8, t0 = m0 & 255;
  v8f acc[8] = {};
  for (int k0 = 0; k0 < CDIM; k0 += 32) {
    int h = k0 >> 8, kc = k0 & 255;
    const __bf16* A = AO + ((((size_t)b * NHEAD + h) * HD + t0) << 8) + kc;
    v16bf a0 = load_A(A, HD, lane);
    v16bf a1 = load_A(A + (size_t)16 * HD, HD, lane);
#pragma unroll
    for (int j = 0; j < 4; ++j) {
      v16bf bt = load_B(Wp + (size_t)(n0 + j * 16) * CDIM + k0, CDIM, lane);
      acc[j]     = __builtin_amdgcn_wmma_f32_16x16x32_bf16(
          false, a0, false, bt, (short)0, acc[j], false, false);
      acc[4 + j] = __builtin_amdgcn_wmma_f32_16x16x32_bf16(
          false, a1, false, bt, (short)0, acc[4 + j], false, false);
    }
  }
  int half = lane >> 4, nl = lane & 15;
#pragma unroll
  for (int i = 0; i < 2; ++i)
#pragma unroll
    for (int j = 0; j < 4; ++j) {
      int o = n0 + j * 16 + nl;
      float bias = bp[o];
#pragma unroll
      for (int r = 0; r < 8; ++r)
        X[(size_t)(m0 + i * 16 + r + (half << 3)) * CDIM + o] =
            acc[i * 4 + j][r] + bias;
    }
}

// ---------------------------------------------------------------------------
extern "C" void kernel_launch(void* const* d_in, const int* in_sizes, int n_in,
                              void* d_out, int out_size, void* d_ws,
                              size_t ws_size, hipStream_t stream) {
  const float* x1 = (const float*)d_in[0];
  const float* x2 = (const float*)d_in[1];
  const float* Wq = (const float*)d_in[2];
  const float* Wk = (const float*)d_in[3];
  const float* Wv = (const float*)d_in[4];
  const float* Wp = (const float*)d_in[5];
  const float* bp = (const float*)d_in[6];

  float* xout = (float*)d_out;                  // [B,N,C] f32
  float* attn = xout + NELEM_X;                 // [B,H,HD,HD] f32

  // workspace carve-up (bytes): 7*64MiB + 4*32MiB = 576 MiB
  char* w = (char*)d_ws;
  const size_t SX = NELEM_X * sizeof(__bf16);   // 64 MiB
  const size_t SW = NELEM_W * sizeof(__bf16);   // 32 MiB
  __bf16* x1b = (__bf16*)w;            w += SX;
  __bf16* x2b = (__bf16*)w;            w += SX;
  __bf16* Wqb = (__bf16*)w;            w += SW;
  __bf16* Wkb = (__bf16*)w;            w += SW;
  __bf16* Wvb = (__bf16*)w;            w += SW;
  __bf16* Wpb = (__bf16*)w;            w += SW;
  __bf16* qT  = (__bf16*)w;            w += SX; // [B,H,HD,N]
  __bf16* kT  = (__bf16*)w;            w += SX;
  __bf16* vT  = (__bf16*)w;            w += SX;
  __bf16* attnb = (__bf16*)w;          w += SX; // bf16 softmax(attn)
  __bf16* aoutb = (__bf16*)w;          w += SX; // attn @ v, bf16

  const int nX = (int)NELEM_X, nW = (int)NELEM_W;
  dim3 tb(256);
  cvt_f32_bf16<<<(nX + 255) / 256, tb, 0, stream>>>(x1, x1b, nX);
  cvt_f32_bf16<<<(nX + 255) / 256, tb, 0, stream>>>(x2, x2b, nX);
  cvt_f32_bf16<<<(nW + 255) / 256, tb, 0, stream>>>(Wq, Wqb, nW);
  cvt_f32_bf16<<<(nW + 255) / 256, tb, 0, stream>>>(Wk, Wkb, nW);
  cvt_f32_bf16<<<(nW + 255) / 256, tb, 0, stream>>>(Wv, Wvb, nW);
  cvt_f32_bf16<<<(nW + 255) / 256, tb, 0, stream>>>(Wp, Wpb, nW);

  dim3 wb(32, 8);  // 8 wave32's per block
  const float SCALE = 1.0f / 16.0f;  // HEAD_DIM^-0.5
  gemm_qkv_T<<<2048, wb, 0, stream>>>(x1b, Wqb, qT, SCALE);
  gemm_qkv_T<<<2048, wb, 0, stream>>>(x2b, Wkb, kT, 1.0f);
  gemm_qkv_T<<<2048, wb, 0, stream>>>(x2b, Wvb, vT, 1.0f);

  gemm_head_logits<<<2048, wb, 0, stream>>>(qT, kT, attn);
  softmax_rows<<<16384, wb, 0, stream>>>(attn, attnb);
  gemm_head_av<<<2048, wb, 0, stream>>>(attnb, vT, aoutb);
  gemm_proj<<<2048, wb, 0, stream>>>(aoutb, Wpb, bp, xout);
}
